// SequentialPolarizedSelfAttention_67877663146634
// MI455X (gfx1250) — matrix operand order repaired and checked by reading, not compiled, not run on hardware
//
#include <hip/hip_runtime.h>
#include <cstdint>
#include <cstddef>

// Problem constants (match reference)
#define B_   16
#define N_   4096
#define C_   512
#define C2_  256
#define ROWS 32
#define NBLK (N_ / ROWS)   // 128 row-blocks per batch

typedef __attribute__((ext_vector_type(16))) _Float16 v16h;
typedef __attribute__((ext_vector_type(8)))  float    v8f;

// ---------------- wave32 reductions ----------------
__device__ inline float wave_sum(float v){
#pragma unroll
  for (int o = 16; o > 0; o >>= 1) v += __shfl_xor(v, o, 32);
  return v;
}
__device__ inline float wave_max(float v){
#pragma unroll
  for (int o = 16; o > 0; o >>= 1) v = fmaxf(v, __shfl_xor(v, o, 32));
  return v;
}

// ---------------- WMMA helpers (CDNA5 16x16x32 f16 -> f32) ----------------
// A (16xK f32 in LDS) gathered per ISA A-layout: lane holds row m=lane&15,
//   halves e<8 -> K = 8*(lane>=16)+e ; e>=8 -> K = 8*(lane>=16)+16+(e-8).
// B gathered per ISA B-layout: lane holds col n=lane&15, same K mapping as A.
// TRANS=1: W stored (K x N) row-major (weight matrices (C,C2)/(C2,C));
// TRANS=0: W stored (N x K) row-major (used for s @ Wv2^T).
template <int TRANS>
__device__ inline v8f gemm_tile(const float* __restrict__ Alds, int ldA,
                                const float* __restrict__ W, int ldW,
                                int n0, int Ktot, v8f acc)
{
  const int lane = threadIdx.x & 31;
  const int m    = lane & 15;
  const int hi   = lane >> 4;
  const int kbA  = hi * 8;
  const int kbB  = hi * 16;
  const int n    = n0 + m;
  for (int k0 = 0; k0 < Ktot; k0 += 32){
    v16h a, b;
#pragma unroll
    for (int e = 0; e < 16; ++e){
      int ka = k0 + kbA + (e < 8 ? e : e + 8);
      a[e] = (_Float16)Alds[m * ldA + ka];
      int kb = k0 + kbA + (e < 8 ? e : e + 8);
      b[e] = (_Float16)(TRANS ? W[(size_t)kb * ldW + n]
                              : W[(size_t)n  * ldW + kb]);
    }
    // (neg_a, A, neg_b, B, c_mod, C, reuse_a, reuse_b)
    acc = __builtin_amdgcn_wmma_f32_16x16x32_f16(false, a, false, b,
                                                 (short)0, acc, false, false);
  }
  return acc;
}

// D layout: VGPR v -> row v + 8*(lane>=16), col = n0 + (lane&15)
__device__ inline void store_tile(float* dst, int ld, int n0, v8f d){
  const int lane = threadIdx.x & 31;
  const int m = lane & 15, hi = lane >> 4;
#pragma unroll
  for (int v = 0; v < 8; ++v) dst[(v + 8 * hi) * ld + n0 + m] = d[v];
}

// ---------------- Pass 1: flash-softmax partials over x ----------------
// Per (batch, 32-row block): q_r = x_r . wq + bq ; block max m, wr = exp(q-m);
// partial weighted col-sum pxw[c] = sum_r wr*x[r,c]; plain col-sum pcs[c].
__global__ __launch_bounds__(256)
void pass1_kernel(const float* __restrict__ x, const float* __restrict__ wq,
                  const float* __restrict__ bq,
                  float* __restrict__ pmax, float* __restrict__ psum,
                  float* __restrict__ pxw,  float* __restrict__ pcs)
{
  const int blk = blockIdx.x;           // 0..NBLK-1
  const int b   = blockIdx.y;           // 0..B-1
  const int tid = threadIdx.x;
  const int lane = tid & 31, wv = tid >> 5;
  const float* xb = x + ((size_t)b * N_ + (size_t)blk * ROWS) * C_;

  __shared__ float wqs[C_];
  __shared__ float qrow[ROWS];
  __shared__ float wr[ROWS];

  for (int i = tid; i < C_; i += 256) wqs[i] = wq[i];
  __syncthreads();
  const float bqv = bq[0];

  // stage A: per-row dots (wave wv owns rows 4wv..4wv+3)
#pragma unroll
  for (int rr = 0; rr < 4; ++rr){
    int r = wv * 4 + rr;
    float acc = 0.f;
    for (int c = lane; c < C_; c += 32) acc += xb[r * C_ + c] * wqs[c];
    acc = wave_sum(acc);
    if (lane == 0) qrow[r] = acc + bqv;
  }
  __syncthreads();

  float m = -1e30f;
#pragma unroll
  for (int r = 0; r < ROWS; ++r) m = fmaxf(m, qrow[r]);
  if (tid < ROWS) wr[tid] = __expf(qrow[tid] - m);
  __syncthreads();

  // stage B: column reductions (x rows are L1/L2 hot from stage A)
#pragma unroll
  for (int cc = 0; cc < 2; ++cc){
    int c = tid + cc * 256;
    float axw = 0.f, acs = 0.f;
#pragma unroll 8
    for (int r = 0; r < ROWS; ++r){
      float v = xb[r * C_ + c];
      acs += v;
      axw += wr[r] * v;
    }
    size_t o = ((size_t)(b * NBLK + blk)) * C_ + c;
    pxw[o] = axw; pcs[o] = acs;
  }
  if (tid == 0){
    float s = 0.f;
#pragma unroll
    for (int r = 0; r < ROWS; ++r) s += wr[r];
    pmax[b * NBLK + blk] = m;
    psum[b * NBLK + blk] = s;
  }
}

// ---------------- Pass 2: merge partials -> xw (softmax-weighted avg), xmean
__global__ __launch_bounds__(256)
void pass2_kernel(const float* __restrict__ pmax, const float* __restrict__ psum,
                  const float* __restrict__ pxw,  const float* __restrict__ pcs,
                  float* __restrict__ xw, float* __restrict__ xmean)
{
  const int b = blockIdx.x, tid = threadIdx.x;
  float m = -1e30f;
  for (int i = 0; i < NBLK; ++i) m = fmaxf(m, pmax[b * NBLK + i]);
  float Z = 0.f;
  for (int i = 0; i < NBLK; ++i) Z += psum[b * NBLK + i] * __expf(pmax[b * NBLK + i] - m);
  const float invZ = 1.f / Z;
#pragma unroll
  for (int cc = 0; cc < 2; ++cc){
    int c = tid + cc * 256;
    float axw = 0.f, acs = 0.f;
    for (int i = 0; i < NBLK; ++i){
      float s = __expf(pmax[b * NBLK + i] - m);
      size_t o = ((size_t)(b * NBLK + i)) * C_ + c;
      axw += pxw[o] * s;
      acs += pcs[o];
    }
    xw[b * C_ + c]    = axw * invZ;
    xmean[b * C_ + c] = acs * (1.f / N_);
  }
}

// ---------------- Head: WMMA chain (single workgroup, 16 waves) -----------
// t1 = xw@Wv1+bv1 ; z = t1@Wz+bz ; cw = sigmoid(LN(z)) ;
// sq = (cw*xmean)@Wq2+bq2 ; s = softmax(sq) ; veff = cw * (s@Wv2^T) ;
// sb = bv2 . s
__global__ __launch_bounds__(512)
void head_kernel(const float* __restrict__ xw_g, const float* __restrict__ xmean_g,
                 const float* __restrict__ Wv1, const float* __restrict__ bv1,
                 const float* __restrict__ Wz,  const float* __restrict__ bz,
                 const float* __restrict__ gamma, const float* __restrict__ beta,
                 const float* __restrict__ Wv2, const float* __restrict__ bv2,
                 const float* __restrict__ Wq2, const float* __restrict__ bq2,
                 float* __restrict__ cw_g, float* __restrict__ veff_g,
                 float* __restrict__ sb_g)
{
  __shared__ float bufA[B_ * C_];   // 32 KB
  __shared__ float bufB[B_ * C_];   // 32 KB
  const int tid = threadIdx.x, lane = tid & 31, wv = tid >> 5;
  const int m16 = lane & 15;

  for (int i = tid; i < B_ * C_; i += 512) bufA[i] = xw_g[i];
  __syncthreads();

  // GEMM1: bufA(16x512) @ Wv1(512x256, trans) -> t1 in bufB (ld 256)
  {
    int n0 = wv * 16;
    v8f t = {};
    t = gemm_tile<1>(bufA, C_, Wv1, C2_, n0, C_, t);
    float bias = bv1[n0 + m16];
#pragma unroll
    for (int v = 0; v < 8; ++v) t[v] += bias;
    store_tile(bufB, C2_, n0, t);
  }
  __syncthreads();

  // GEMM2: t1(16x256) @ Wz(256x512, trans) -> z in bufA (ld 512)
#pragma unroll
  for (int t2 = 0; t2 < 2; ++t2){
    int n0 = wv * 32 + t2 * 16;
    v8f z = {};
    z = gemm_tile<1>(bufB, C2_, Wz, C_, n0, C2_, z);
    float bias = bz[n0 + m16];
#pragma unroll
    for (int v = 0; v < 8; ++v) z[v] += bias;
    store_tile(bufA, C_, n0, z);
  }
  __syncthreads();

  // LayerNorm(eps=1e-3) + sigmoid, wave wv owns batch row wv -> cw in bufA
  {
    float* row = bufA + wv * C_;
    float s = 0.f, sq = 0.f;
#pragma unroll
    for (int j = 0; j < 16; ++j){ float v = row[lane + 32 * j]; s += v; sq += v * v; }
    s = wave_sum(s); sq = wave_sum(sq);
    float mean = s * (1.f / C_);
    float var  = sq * (1.f / C_) - mean * mean;
    float inv  = rsqrtf(var + 1e-3f);
#pragma unroll
    for (int j = 0; j < 16; ++j){
      int c = lane + 32 * j;
      float v = (row[c] - mean) * inv * gamma[c] + beta[c];
      row[c] = 1.f / (1.f + __expf(-v));
    }
  }
  __syncthreads();

  // g = cw * xmean -> bufB (16x512)
  for (int i = tid; i < B_ * C_; i += 512) bufB[i] = bufA[i] * xmean_g[i];
  __syncthreads();

  // GEMM3: g(16x512) @ Wq2(512x256, trans) -> sq ; regs until barrier, then bufB
  {
    int n0 = wv * 16;
    v8f t = {};
    t = gemm_tile<1>(bufB, C_, Wq2, C2_, n0, C_, t);
    float bias = bq2[n0 + m16];
#pragma unroll
    for (int v = 0; v < 8; ++v) t[v] += bias;
    __syncthreads();                 // all waves done reading g
    store_tile(bufB, C2_, n0, t);
  }
  __syncthreads();

  // softmax over 256 per row + sb = bv2 . s
  {
    float* row = bufB + wv * C2_;
    float mx = -1e30f;
#pragma unroll
    for (int j = 0; j < 8; ++j) mx = fmaxf(mx, row[lane + 32 * j]);
    mx = wave_max(mx);
    float e[8]; float s = 0.f;
#pragma unroll
    for (int j = 0; j < 8; ++j){ e[j] = __expf(row[lane + 32 * j] - mx); s += e[j]; }
    s = wave_sum(s);
    float inv = 1.f / s;
    float sbacc = 0.f;
#pragma unroll
    for (int j = 0; j < 8; ++j){
      int c = lane + 32 * j;
      float sv = e[j] * inv;
      row[c] = sv;
      sbacc += sv * bv2[c];
    }
    sbacc = wave_sum(sbacc);
    if (lane == 0) sb_g[wv] = sbacc;
  }
  __syncthreads();

  // GEMM4: s(16x256) @ Wv2^T (Wv2 is (512,256), no-trans) -> veff_raw in bufB (ld 512)
  {
    int n00 = wv * 32, n01 = n00 + 16;
    v8f t0 = {}, t1 = {};
    t0 = gemm_tile<0>(bufB, C2_, Wv2, C2_, n00, C2_, t0);
    t1 = gemm_tile<0>(bufB, C2_, Wv2, C2_, n01, C2_, t1);
    __syncthreads();                 // all waves done reading s
    store_tile(bufB, C_, n00, t0);
    store_tile(bufB, C_, n01, t1);
  }
  __syncthreads();

  for (int i = tid; i < B_ * C_; i += 512){
    float cw = bufA[i];
    cw_g[i]   = cw;
    veff_g[i] = cw * bufB[i];
  }
}

// ---------------- Pass 3: out = sigmoid(x.veff + sb) * cw * x --------------
__global__ __launch_bounds__(256)
void pass3_kernel(const float* __restrict__ x,
                  const float* __restrict__ cw_g, const float* __restrict__ veff_g,
                  const float* __restrict__ sb_g, float* __restrict__ out)
{
  const int blk = blockIdx.x, b = blockIdx.y;
  const int tid = threadIdx.x, lane = tid & 31, wv = tid >> 5;
  const float* xb = x   + ((size_t)b * N_ + (size_t)blk * ROWS) * C_;
  float*       ob = out + ((size_t)b * N_ + (size_t)blk * ROWS) * C_;
  __shared__ float veffs[C_], cws[C_], srow[ROWS];

  for (int i = tid; i < C_; i += 256){
    veffs[i] = veff_g[b * C_ + i];
    cws[i]   = cw_g[b * C_ + i];
  }
  __syncthreads();
  const float sbv = sb_g[b];

#pragma unroll
  for (int rr = 0; rr < 4; ++rr){
    int r = wv * 4 + rr;
    float acc = 0.f;
    for (int c = lane; c < C_; c += 32) acc += xb[r * C_ + c] * veffs[c];
    acc = wave_sum(acc);
    if (lane == 0) srow[r] = 1.f / (1.f + __expf(-(acc + sbv)));
  }
  __syncthreads();

#pragma unroll
  for (int cc = 0; cc < 2; ++cc){
    int c = tid + cc * 256;
    float cwv = cws[c];
    for (int r = 0; r < ROWS; ++r)
      ob[r * C_ + c] = srow[r] * cwv * xb[r * C_ + c];
  }
}

// ---------------- launcher ----------------
extern "C" void kernel_launch(void* const* d_in, const int* in_sizes, int n_in,
                              void* d_out, int out_size, void* d_ws, size_t ws_size,
                              hipStream_t stream)
{
  (void)in_sizes; (void)n_in; (void)out_size; (void)ws_size;
  const float* x     = (const float*)d_in[0];
  const float* Wv1   = (const float*)d_in[1];   // ch_wv_w (512,256)
  const float* bv1   = (const float*)d_in[2];
  const float* wq    = (const float*)d_in[3];   // ch_wq_w (512,1)
  const float* bq    = (const float*)d_in[4];
  const float* Wz    = (const float*)d_in[5];   // ch_wz_w (256,512)
  const float* bz    = (const float*)d_in[6];
  const float* gamma = (const float*)d_in[7];
  const float* beta  = (const float*)d_in[8];
  const float* Wv2   = (const float*)d_in[9];   // sp_wv_w (512,256)
  const float* bv2   = (const float*)d_in[10];
  const float* Wq2   = (const float*)d_in[11];  // sp_wq_w (512,256)
  const float* bq2   = (const float*)d_in[12];
  float* out = (float*)d_out;

  // workspace layout (floats), ~8.6 MB, fully overwritten every call
  float* ws    = (float*)d_ws;
  float* pmax  = ws;
  float* psum  = pmax + B_ * NBLK;
  float* pxw   = psum + B_ * NBLK;
  float* pcs   = pxw + (size_t)B_ * NBLK * C_;
  float* xw    = pcs + (size_t)B_ * NBLK * C_;
  float* xmean = xw + B_ * C_;
  float* cw    = xmean + B_ * C_;
  float* veff  = cw + B_ * C_;
  float* sb    = veff + B_ * C_;

  dim3 grid(NBLK, B_);
  pass1_kernel<<<grid, 256, 0, stream>>>(x, wq, bq, pmax, psum, pxw, pcs);
  pass2_kernel<<<B_, 256, 0, stream>>>(pmax, psum, pxw, pcs, xw, xmean);
  head_kernel<<<1, 512, 0, stream>>>(xw, xmean, Wv1, bv1, Wz, bz, gamma, beta,
                                     Wv2, bv2, Wq2, bq2, cw, veff, sb);
  pass3_kernel<<<grid, 256, 0, stream>>>(x, cw, veff, sb, out);
}
